// ParticleFilter_48155173322874
// MI455X (gfx1250) — matrix operand order repaired and checked by reading, not compiled, not run on hardware
//
#include <hip/hip_runtime.h>

// Particle filter for MI455X (gfx1250): one workgroup per batch, full T-loop
// in-kernel, particles ping-ponged in global scratch (fits in 192MB L2).
// Likelihood reduction Σ_d (p-obs)^2 done as diag(Diff·Diff^T) with
// v_wmma_f32_16x16x32_f16 per 16-particle tile.

typedef __attribute__((ext_vector_type(16))) _Float16 v16h;
typedef __attribute__((ext_vector_type(8)))  float    v8f;

#define NPART   2048
#define DDIM    128
#define TSTEPS  64
#define BLOCK   256
#define NWAVES  8                      // 256 / wave32
#define TILES_PER_WAVE 16              // (NPART/16) tiles / NWAVES
#define PER_THREAD (NPART / BLOCK)     // 8 particles per thread

__device__ __forceinline__ unsigned hash_u32(unsigned x) {
  x ^= x >> 17; x *= 0xed5ad4bbu;
  x ^= x >> 11; x *= 0xac4c1b51u;
  x ^= x >> 15; x *= 0x31848babu;
  x ^= x >> 14; return x;
}
__device__ __forceinline__ float u01(unsigned h) {
  return (float)(h >> 8) * (1.0f / 16777216.0f);
}

__global__ __launch_bounds__(BLOCK)
void pf_persistent_kernel(const float* __restrict__ z,
                          const float* __restrict__ obs,   // [B, D, T]
                          float* __restrict__ out,         // [B, D]
                          float* __restrict__ pA,
                          float* __restrict__ pB)
{
  __shared__ _Float16 s_diff[NWAVES][16 * DDIM];  // 32 KB: per-wave 16xD f16 tile
  __shared__ float    s_w[NPART];                 // 8 KB  persistent weights
  __shared__ float    s_lik[NPART];               // 8 KB
  __shared__ float    s_cdf[NPART];               // 8 KB
  __shared__ float    s_part[BLOCK];              // 1 KB  scan partials
  __shared__ float    s_obs[DDIM];                // 512 B

  const int b    = blockIdx.x;
  const int tid  = threadIdx.x;
  const int lane = tid & 31;
  const int wave = tid >> 5;

  float* pcur = pA + (size_t)b * NPART * DDIM;
  float* pnxt = pB + (size_t)b * NPART * DDIM;

  // init: particles = broadcast z[b,:], weights = 1/N
  for (int e = tid; e < NPART * DDIM; e += BLOCK)
    pcur[e] = z[b * DDIM + (e & (DDIM - 1))];
  for (int i = tid; i < NPART; i += BLOCK)
    s_w[i] = 1.0f / (float)NPART;
  __syncthreads();

  for (int t = 0; t < TSTEPS; ++t) {
    // stage obs[b,:,t]; prefetch next timestep's column into GL2
    if (tid < DDIM) {
      s_obs[tid] = obs[(b * DDIM + tid) * TSTEPS + t];
      if (t + 1 < TSTEPS)
        __builtin_prefetch(&obs[(b * DDIM + tid) * TSTEPS + t + 1], 0, 0);
    }
    __syncthreads();

    // ---- diffusion noise + likelihood via WMMA Gram diagonal ----
    for (int iter = 0; iter < TILES_PER_WAVE; ++iter) {
      const int tile = iter * NWAVES + wave;   // 0..127
      const int p0   = tile * 16;

      // noise + f16 diff staging for this wave's 16-particle tile
      for (int e = lane; e < 16 * DDIM; e += 32) {
        const int m  = e >> 7;                 // particle within tile
        const int d  = e & (DDIM - 1);
        const int gp = p0 + m;
        const size_t off = (size_t)gp * DDIM + d;
        float p = pcur[off];
        const unsigned seed =
            (((unsigned)(b * TSTEPS + t) * (unsigned)NPART + (unsigned)gp)
             * (unsigned)DDIM + (unsigned)d);
        const float uu1 = u01(hash_u32(seed * 2u + 0x68bc21ebu)) + 1e-7f;
        const float uu2 = u01(hash_u32(seed * 2u + 0x02e5be93u));
        const float r = __fsqrt_rn(-2.0f * __logf(uu1));
        p += 0.1f * (r * __cosf(6.28318530718f * uu2));
        pcur[off] = p;
        s_diff[wave][(m << 7) + d] = (_Float16)(p - s_obs[d]);
      }
      __syncthreads();

      // Gram tile: C += A(16x32) * B(32x16) over 4 K-chunks, B = Diff^T
      v8f c = {};
      const int row = lane & 15;
      const int kA  = (lane < 16) ? 0 : 8;    // A-layout K base (16-bit A 16x32)
      const int kB  = (lane < 16) ? 0 : 8;    // B K base, same table order as A
      #pragma unroll
      for (int kk = 0; kk < DDIM; kk += 32) {
        union { v16h v; _Float16 h[16]; } ua, ub;
        #pragma unroll
        for (int j = 0; j < 16; ++j) {
          const int ka = kk + ((j < 8) ? (kA + j) : (16 + kA + (j - 8)));
          ua.h[j] = s_diff[wave][(row << 7) + ka];       // A[m=row][ka]
        }
        #pragma unroll
        for (int j = 0; j < 16; ++j) {
          const int kb = kk + ((j < 8) ? (kB + j) : (16 + kB + (j - 8)));
          ub.h[j] = s_diff[wave][(row << 7) + kb];       // B[kb][n=row] = Diff[n][kb]
        }
        c = __builtin_amdgcn_wmma_f32_16x16x32_f16(
                false, ua.v, false, ub.v, (short)0, c, false, false);
      }

      // diagonal extraction per 16x16 f32 C layout
      float ss = 0.0f; int m = -1;
      if (lane < 8)        { m = lane;       ss = c[lane];       }
      else if (lane >= 24) { m = lane - 16;  ss = c[lane - 24];  }
      if (m >= 0)
        s_lik[p0 + m] = __expf(-0.5f * ss) + 1e-8f;
      __syncthreads();
    }

    // ---- weight update + inclusive scan (CDF) ----
    const int j0 = tid * PER_THREAD;
    float loc[PER_THREAD];
    float run = 0.0f;
    #pragma unroll
    for (int k = 0; k < PER_THREAD; ++k) {
      const float w = s_w[j0 + k] * s_lik[j0 + k] + 1e-10f;
      run += w;
      loc[k] = run;            // local inclusive prefix
      s_w[j0 + k] = w;         // unnormalized; normalized below
    }
    s_part[tid] = run;
    __syncthreads();
    #pragma unroll
    for (int off = 1; off < BLOCK; off <<= 1) {
      const float v = (tid >= off) ? s_part[tid - off] : 0.0f;
      __syncthreads();
      s_part[tid] += v;
      __syncthreads();
    }
    const float total    = s_part[BLOCK - 1];
    const float inv_tot  = 1.0f / total;
    const float excl     = (tid > 0) ? s_part[tid - 1] : 0.0f;
    #pragma unroll
    for (int k = 0; k < PER_THREAD; ++k) {
      s_cdf[j0 + k] = excl + loc[k];
      s_w[j0 + k]  *= inv_tot;          // persist normalized weights
    }
    __syncthreads();

    // ---- multinomial resampling: inverse-CDF + float4 gather ----
    for (int k = 0; k < PER_THREAD; ++k) {
      const int j = j0 + k;
      const unsigned hs =
          hash_u32(((unsigned)(b * TSTEPS + t) * (unsigned)NPART + (unsigned)j)
                   ^ 0x9e3779b9u);
      const float u = u01(hs) * total;
      int lo = 0, hi = NPART - 1;
      while (lo < hi) {
        const int mid = (lo + hi) >> 1;
        if (s_cdf[mid] < u) lo = mid + 1; else hi = mid;
      }
      const float4* src = (const float4*)(pcur + (size_t)lo * DDIM);
      float4*       dst = (float4*)(pnxt + (size_t)j * DDIM);
      #pragma unroll 8
      for (int q = 0; q < DDIM / 4; ++q) dst[q] = src[q];
    }
    __syncthreads();
    float* tmp = pcur; pcur = pnxt; pnxt = tmp;
  }

  // ---- mean over particles ----
  if (tid < DDIM) {
    float acc = 0.0f;
    for (int i = 0; i < NPART; ++i)
      acc += pcur[(size_t)i * DDIM + tid];
    out[b * DDIM + tid] = acc * (1.0f / (float)NPART);
  }
}

extern "C" void kernel_launch(void* const* d_in, const int* in_sizes, int n_in,
                              void* d_out, int out_size, void* d_ws, size_t ws_size,
                              hipStream_t stream) {
  const float* z   = (const float*)d_in[0];           // [B, 128]
  const float* obs = (const float*)d_in[1];           // [B, 128, 64]
  float* out = (float*)d_out;                         // [B, 128]

  const int B = in_sizes[0] / DDIM;                   // 64
  const size_t buf_elems = (size_t)B * NPART * DDIM;  // 16M floats

  float* pA = (float*)d_ws;
  // ping-pong buffer; degrade to in-place if scratch is too small
  float* pB = (ws_size >= 2 * buf_elems * sizeof(float)) ? (pA + buf_elems) : pA;

  pf_persistent_kernel<<<B, BLOCK, 0, stream>>>(z, obs, out, pA, pB);
}
